// CapsNetBasic_41489384079951
// MI455X (gfx1250) — matrix operand order repaired and checked by reading, not compiled, not run on hardware
//
#include <hip/hip_runtime.h>
#include <hip/hip_bf16.h>

#define HW   256
#define NPIX (HW*HW)
#define CAPS_EPS 1e-8f

typedef __attribute__((ext_vector_type(16))) _Float16 v16h;
typedef __attribute__((ext_vector_type(8)))  _Float16 v8h;
typedef __attribute__((ext_vector_type(8)))  float    v8f;
typedef __attribute__((ext_vector_type(4)))  float    f4;

// ---------------------------------------------------------------------------
// K0: weight prep — wp[256][128][5][5] f32  ->  wpt[256][25][128] f16
// (channel-innermost so WMMA B fragments are contiguous 32B per lane)
// ---------------------------------------------------------------------------
__global__ void k_prep_wp(const float* __restrict__ wp, _Float16* __restrict__ wpt) {
    int idx = blockIdx.x * 256 + threadIdx.x;           // over 256*25*128
    if (idx >= 256 * 25 * 128) return;
    int c   = idx % 128;
    int tap = (idx / 128) % 25;
    int n   = idx / (128 * 25);
    wpt[idx] = (_Float16)wp[(n * 128 + c) * 25 + tap];
}

// ---------------------------------------------------------------------------
// K1: conv1 1->128, 5x5 SAME.  Writes hT[pixel][128] in f16 (WMMA A operand).
// ---------------------------------------------------------------------------
__global__ void k_conv1(const float* __restrict__ x, const float* __restrict__ w1,
                        const float* __restrict__ b1, _Float16* __restrict__ hT) {
    __shared__ float w[128 * 25];
    __shared__ float b[128];
    for (int i = threadIdx.x; i < 128 * 25; i += 256) w[i] = w1[i];
    for (int i = threadIdx.x; i < 128;      i += 256) b[i] = b1[i];
    __syncthreads();

    int p  = blockIdx.x * 256 + threadIdx.x;            // one pixel per thread
    int py = p >> 8, px = p & 255;

    float xv[25];
#pragma unroll
    for (int t = 0; t < 25; ++t) {
        int dy = t / 5 - 2, dx = t % 5 - 2;
        int yy = py + dy, xx = px + dx;
        xv[t] = (yy >= 0 && yy < HW && xx >= 0 && xx < HW) ? x[yy * HW + xx] : 0.0f;
    }
    for (int c = 0; c < 128; ++c) {
        float acc = b[c];
#pragma unroll
        for (int t = 0; t < 25; ++t) acc += w[c * 25 + t] * xv[t];
        hT[p * 128 + c] = (_Float16)acc;
    }
}

// ---------------------------------------------------------------------------
// K2: primary-caps conv 128->256, 5x5 SAME as implicit GEMM on WMMA.
//
// Block = 256 thr (8 waves) covers a 64-pixel row window x 256 channels.
// A halo tile (5 rows x 68 cols x 128ch f16 = 87KB) staged in LDS once
// (zero-padded) -> branch-free tap loop, A fragments via ds_load_b128.
// Wave w handles a 32-channel slice over all 64 pixels:
//   4 M-subtiles x 2 N-tiles (64 accumulator VGPRs), K stepped by 32 with
//   v_wmma_f32_16x16x32_f16 (fp32 accumulation).
// B fragments loaded per k-step (16 live VGPRs) — scheduler pipelines them;
// __launch_bounds__(256,2) raises the VGPR budget so nothing spills.
// 1024 blocks => B (weight) L2 traffic = 1024 x 1.6MB ~= 1.65 GB total.
// ---------------------------------------------------------------------------
#define XW  64            // pixels per block
#define XWH (XW + 4)      // halo width = 68

__global__ void __launch_bounds__(256, 2)
k_conv2_wmma(const _Float16* __restrict__ hT,
             const _Float16* __restrict__ wpt,
             const float*    __restrict__ bp,
             float*          __restrict__ votes) {
    __shared__ _Float16 sA[5 * XWH * 128];              // 87,040 B (WGP has 320KB)

    const int tid = threadIdx.x;
    const int y   = blockIdx.x >> 2;                    // 256 rows
    const int xb0 = (blockIdx.x & 3) * XW;              // 4 windows/row

    // ---- stage A tile: rows y-2..y+2, cols xb0-2..xb0+65, 128 ch (zero pad)
    for (int j = tid; j < 5 * XWH * 16; j += 256) {     // 16B chunks
        const int c8   = j & 15;
        const int xcol = (j >> 4) % XWH;
        const int r    = j / (16 * XWH);
        const int gy = y + r - 2, gx = xb0 + xcol - 2;
        v8h val = {};
        if (gy >= 0 && gy < HW && gx >= 0 && gx < HW)
            val = *(const v8h*)(hT + (size_t)(gy * HW + gx) * 128 + c8 * 8);
        *(v8h*)(sA + ((r * XWH + xcol) * 128 + c8 * 8)) = val;
    }
    __syncthreads();

    const int lane  = tid & 31;
    const int wv    = tid >> 5;                         // 0..7 -> 32-ch slice
    const int nbase = wv * 32;
    const int mrow  = lane & 15;                        // A: lane -> M row
    const int hi    = lane >> 4;                        // half-wave selector
    const int nloc  = lane & 15;                        // B/C/D: lane -> N col

    v8f acc[4][2] = {};                                 // [mt][nt]

    // per-lane B base pointers (n = nbase + nt*16 + nloc), + tap*128 + kb*32
    const _Float16* bp0 = wpt + ((size_t)(nbase + nloc)      * 25) * 128 + hi * 16;
    const _Float16* bp1 = wpt + ((size_t)(nbase + 16 + nloc) * 25) * 128 + hi * 16;

    for (int tap = 0; tap < 25; ++tap) {
        const int dy = tap / 5, dx = tap % 5;           // halo-indexed 0..4
        const int abase = (dy * XWH + dx) * 128;

        // prefetch next tap's weight lines into near caches while WMMAs run
        if (tap < 24) {
            __builtin_prefetch(bp0 + (tap + 1) * 128, 0, 3);
            __builtin_prefetch(bp1 + (tap + 1) * 128, 0, 3);
        }

#pragma unroll
        for (int kb = 0; kb < 4; ++kb) {
            // --- B fragments for this k-step (2 n-tiles), 16 live VGPRs
            const int koff = tap * 128 + kb * 32;
            const v16h b0 = *(const v16h*)(bp0 + koff);
            const v16h b1 = *(const v16h*)(bp1 + koff);

#pragma unroll
            for (int mt = 0; mt < 4; ++mt) {
                // A fragment from LDS: lanes hold M=lane%16;
                // VGPR0-7 = K (hi?16..31:0..15)
                const int lb = abase + (mt * 16 + mrow) * 128 + kb * 32 + hi * 16;
                v8h c1 = *(const v8h*)(sA + lb);
                v8h c2 = *(const v8h*)(sA + lb + 8);
                v16h a;
#pragma unroll
                for (int i = 0; i < 8; ++i) { a[i] = c1[i]; a[8 + i] = c2[i]; }

                acc[mt][0] = __builtin_amdgcn_wmma_f32_16x16x32_f16(
                    false, a, false, b0, (short)0, acc[mt][0], false, false);
                acc[mt][1] = __builtin_amdgcn_wmma_f32_16x16x32_f16(
                    false, a, false, b1, (short)0, acc[mt][1], false, false);
            }
        }
    }

    // Epilogue: C/D layout => lane holds N=lane%16, M = vgpr + 8*hi
#pragma unroll
    for (int nt = 0; nt < 2; ++nt) {
        const int   n    = nbase + nt * 16 + nloc;
        const float bias = bp[n];
#pragma unroll
        for (int mt = 0; mt < 4; ++mt) {
#pragma unroll
            for (int r = 0; r < 8; ++r) {
                const int pix = y * HW + xb0 + mt * 16 + hi * 8 + r;
                votes[(size_t)pix * 256 + n] = acc[mt][nt][r] + bias;
            }
        }
    }
}

// ---------------------------------------------------------------------------
// K3: both routings collapsed (routing#1: c==1/32 since b==0 & iters==1;
// routing#2: softmax over size-1 axis == 1, so all 5 iters are identical):
//   v1[i]  = squash(votes[i]/32)  over 8 atoms
//   U[a2]  = sum_i sum_a ws[a2][a]*v1[i][a] + 32*bs[a2]
//   v2     = squash(U) over 16 atoms
// ---------------------------------------------------------------------------
__global__ void k_route_seg(const float* __restrict__ votes,
                            const float* __restrict__ wsW,
                            const float* __restrict__ bs,
                            float* __restrict__ v2) {
    __shared__ float w[16 * 8];
    if (threadIdx.x < 128) w[threadIdx.x] = wsW[threadIdx.x];
    __syncthreads();

    int p = blockIdx.x * 256 + threadIdx.x;             // one pixel per thread
    float U[16];
#pragma unroll
    for (int a2 = 0; a2 < 16; ++a2) U[a2] = 32.0f * bs[a2];

    const float* vp = votes + (size_t)p * 256;
    for (int i = 0; i < 32; ++i) {
        f4 s0 = *(const f4*)(vp + i * 8);
        f4 s1 = *(const f4*)(vp + i * 8 + 4);
        const float inv = 1.0f / 32.0f;
        float s[8] = { s0.x * inv, s0.y * inv, s0.z * inv, s0.w * inv,
                       s1.x * inv, s1.y * inv, s1.z * inv, s1.w * inv };
        float sq = 0.f;
#pragma unroll
        for (int a = 0; a < 8; ++a) sq += s[a] * s[a];
        float coef = (sq / (1.0f + sq)) / sqrtf(sq + CAPS_EPS);
#pragma unroll
        for (int a2 = 0; a2 < 16; ++a2) {
            float acc = 0.f;
#pragma unroll
            for (int a = 0; a < 8; ++a) acc += w[a2 * 8 + a] * s[a];
            U[a2] += coef * acc;
        }
    }
    float sq2 = 0.f;
#pragma unroll
    for (int a2 = 0; a2 < 16; ++a2) sq2 += U[a2] * U[a2];
    float coef2 = (sq2 / (1.0f + sq2)) / sqrtf(sq2 + CAPS_EPS);
#pragma unroll
    for (int a2 = 0; a2 < 16; ++a2) v2[(size_t)p * 16 + a2] = coef2 * U[a2];
}

// ---------------------------------------------------------------------------
// K4: final conv 16->10, 5x5 SAME; emits (v_lens = |x_out|, x_out).
// ---------------------------------------------------------------------------
__global__ void k_final(const float* __restrict__ v2,
                        const float* __restrict__ wr,
                        const float* __restrict__ br,
                        float* __restrict__ out) {
    __shared__ float w[10 * 16 * 25];                   // 16KB
    __shared__ float b[10];
    for (int i = threadIdx.x; i < 10 * 16 * 25; i += 256) w[i] = wr[i];
    if (threadIdx.x < 10) b[threadIdx.x] = br[threadIdx.x];
    __syncthreads();

    int p  = blockIdx.x * 256 + threadIdx.x;
    int py = p >> 8, px = p & 255;

    float o[10];
#pragma unroll
    for (int c = 0; c < 10; ++c) o[c] = b[c];

    for (int t = 0; t < 25; ++t) {
        int dy = t / 5 - 2, dx = t % 5 - 2;
        int yy = py + dy, xx = px + dx;
        if (yy < 0 || yy >= HW || xx < 0 || xx >= HW) continue;
        const float* vp = v2 + (size_t)(yy * HW + xx) * 16;
        float val[16];
#pragma unroll
        for (int a = 0; a < 16; ++a) val[a] = vp[a];
#pragma unroll
        for (int c = 0; c < 10; ++c) {
            float acc = 0.f;
#pragma unroll
            for (int a = 0; a < 16; ++a) acc += w[(c * 16 + a) * 25 + t] * val[a];
            o[c] += acc;
        }
    }
#pragma unroll
    for (int c = 0; c < 10; ++c) {
        float v = o[c];
        out[c * NPIX + p]          = fabsf(v);          // v_lens
        out[655360 + c * NPIX + p] = v;                 // x_out
    }
}

// ---------------------------------------------------------------------------
// launch — inputs (dict order): x, w1, b1, wp, bp, ws, bs, wr, br
// ---------------------------------------------------------------------------
extern "C" void kernel_launch(void* const* d_in, const int* in_sizes, int n_in,
                              void* d_out, int out_size, void* d_ws, size_t ws_size,
                              hipStream_t stream) {
    const float* x   = (const float*)d_in[0];
    const float* w1  = (const float*)d_in[1];
    const float* b1  = (const float*)d_in[2];
    const float* wp  = (const float*)d_in[3];
    const float* bp  = (const float*)d_in[4];
    const float* wsW = (const float*)d_in[5];
    const float* bs  = (const float*)d_in[6];
    const float* wr  = (const float*)d_in[7];
    const float* br  = (const float*)d_in[8];

    // workspace carve-up (all offsets 256B-aligned)
    char* base = (char*)d_ws;
    _Float16* hT    = (_Float16*)(base);                             // 16,777,216 B
    _Float16* wpt   = (_Float16*)(base + 16777216);                  //  1,638,400 B
    float*    votes = (float*)   (base + 16777216 + 1638400);        // 67,108,864 B
    float*    v2    = (float*)   (base + 16777216 + 1638400 + 67108864); // 4,194,304 B

    k_prep_wp   <<<3200, 256, 0, stream>>>(wp, wpt);
    k_conv1     <<<NPIX / 256, 256, 0, stream>>>(x, w1, b1, hT);
    k_conv2_wmma<<<1024, 256, 0, stream>>>(hT, wpt, bp, votes);
    k_route_seg <<<NPIX / 256, 256, 0, stream>>>(votes, wsW, bs, v2);
    k_final     <<<NPIX / 256, 256, 0, stream>>>(v2, wr, br, (float*)d_out);
}
